// MambaBackbone_61804579389946
// MI455X (gfx1250) — compile-verified, matrix operand order checked by reading, not run on hardware
//
#include <hip/hip_runtime.h>
#include <hip/hip_bf16.h>

#define D_MODEL   1024
#define D_STATE   64
#define D_INNER   2048
#define NHEADS    32
#define HEADDIM   64
#define CONV_DIM  2176
#define D_IN_PROJ 4256
#define D_CONV    4
#define NLAYERS   2
#define BATCH     2
#define SEQLEN    4096
#define NTOK      (BATCH * SEQLEN)
#define QCHUNK    64
#define NCHUNK    (SEQLEN / QCHUNK)          // 64 chunks per sequence
#define NTILES    (BATCH * NCHUNK * NHEADS)  // 4096 (b,c,h) tiles
#define EPS_F     1e-5f

typedef __bf16 bf16_t;
typedef bf16_t v8bf  __attribute__((ext_vector_type(8)));
typedef bf16_t v16bf __attribute__((ext_vector_type(16)));
typedef float  v8f   __attribute__((ext_vector_type(8)));

// ---------------------------------------------------------------------------
// CDNA5 async global->LDS copy (16B per lane, tracked by ASYNCcnt).
// lds_off: byte offset within the workgroup's LDS allocation (low 32 bits of a
// generic LDS pointer per the flat-aperture mapping).
// ---------------------------------------------------------------------------
__device__ __forceinline__ void async_load_b128(unsigned lds_off,
                                                const void* gaddr) {
    asm volatile("global_load_async_to_lds_b128 %0, %1, off"
                 :
                 : "v"(lds_off), "v"(gaddr)
                 : "memory");
}
__device__ __forceinline__ void wait_asynccnt0() {
    asm volatile("s_wait_asynccnt 0x0" ::: "memory");
}
__device__ __forceinline__ unsigned lds_off_of(const void* p) {
    return (unsigned)(unsigned long long)p;
}

// ---------------------------------------------------------------------------
// WMMA fragment helpers.  bf16 16x32 fragment: lanes 0-15 hold K {0..7,16..23},
// lanes 16-31 hold K {8..15,24..31}; caller passes chunk base (+ksel applied).
// ---------------------------------------------------------------------------
__device__ __forceinline__ v16bf load_frag(const bf16_t* p) {
    v8bf lo = *(const v8bf*)p;
    v8bf hi = *(const v8bf*)(p + 16);
    return __builtin_shufflevector(lo, hi, 0, 1, 2, 3, 4, 5, 6, 7, 8, 9, 10,
                                   11, 12, 13, 14, 15);
}

// 64x64x64 GEMM on LDS tiles (A: 64x64 row-major, Bt: 64x64 row-major = B^T).
// 8 waves: wave computes a 16x32 slice -> acc[2] 16x16 tiles.
__device__ __forceinline__ void gemm64_accum(const bf16_t* sA, const bf16_t* sBt,
                                             int lane, int wid, v8f acc[2]) {
    const int mrow = (wid >> 1) * 16 + (lane & 15);
    const int nbase = (wid & 1) * 32;
    const int ksel = (lane & 16) ? 8 : 0;
#pragma unroll
    for (int k0 = 0; k0 < 64; k0 += 32) {
        v16bf a = load_frag(&sA[mrow * 64 + k0 + ksel]);
#pragma unroll
        for (int j = 0; j < 2; j++) {
            v16bf b = load_frag(&sBt[(nbase + j * 16 + (lane & 15)) * 64 + k0 + ksel]);
            acc[j] = __builtin_amdgcn_wmma_f32_16x16x32_bf16(
                false, a, false, b, (short)0, acc[j], false, false);
        }
    }
}

// ---------------------------------------------------------------------------
// fp32 -> bf16 conversion (weights / one-off)
// ---------------------------------------------------------------------------
__global__ void f32_to_bf16_kernel(const float* __restrict__ src,
                                   bf16_t* __restrict__ dst, int n) {
    int i = blockIdx.x * blockDim.x + threadIdx.x;
    int stride = gridDim.x * blockDim.x;
    for (; i < n; i += stride) dst[i] = (bf16_t)src[i];
}

// ---------------------------------------------------------------------------
// LayerNorm (fp32 in) -> bf16 out.  One block per token.
// ---------------------------------------------------------------------------
__global__ __launch_bounds__(256) void layernorm_bf16_kernel(
    const float* __restrict__ in, const float* __restrict__ w,
    const float* __restrict__ b, bf16_t* __restrict__ out) {
    __shared__ float sred[8];
    __shared__ float smean, srstd;
    const int t = blockIdx.x;
    const float* row = in + (size_t)t * D_MODEL;
    float v[4];
    float s = 0.f;
#pragma unroll
    for (int i = 0; i < 4; i++) {
        v[i] = row[threadIdx.x + 256 * i];
        s += v[i];
    }
#pragma unroll
    for (int o = 16; o > 0; o >>= 1) s += __shfl_down(s, o);
    if ((threadIdx.x & 31) == 0) sred[threadIdx.x >> 5] = s;
    __syncthreads();
    if (threadIdx.x == 0) {
        float tot = 0.f;
        for (int i = 0; i < 8; i++) tot += sred[i];
        smean = tot * (1.f / D_MODEL);
    }
    __syncthreads();
    const float m = smean;
    float q = 0.f;
#pragma unroll
    for (int i = 0; i < 4; i++) {
        float d = v[i] - m;
        q += d * d;
    }
#pragma unroll
    for (int o = 16; o > 0; o >>= 1) q += __shfl_down(q, o);
    if ((threadIdx.x & 31) == 0) sred[threadIdx.x >> 5] = q;
    __syncthreads();
    if (threadIdx.x == 0) {
        float tot = 0.f;
        for (int i = 0; i < 8; i++) tot += sred[i];
        srstd = rsqrtf(tot * (1.f / D_MODEL) + EPS_F);
    }
    __syncthreads();
    const float rs = srstd;
#pragma unroll
    for (int i = 0; i < 4; i++) {
        int j = threadIdx.x + 256 * i;
        out[(size_t)t * D_MODEL + j] = (bf16_t)((v[i] - m) * rs * w[j] + b[j]);
    }
}

// ---------------------------------------------------------------------------
// bf16 WMMA GEMM:  C[MxN] (+ optional residual) = A[MxK] * Bt[NxK]^T
// Block tile 128x128, 8 waves in a 4x2 grid, each wave does a 32x64 tile
// (2x4 of v_wmma_f32_16x16x32_bf16).  Staging via async global->LDS DMA.
// ---------------------------------------------------------------------------
__global__ __launch_bounds__(256) void gemm_bf16_wmma_kernel(
    const bf16_t* __restrict__ A,   // M x K, row-major, bf16
    const bf16_t* __restrict__ Bt,  // N x K, row-major (i.e. B transposed)
    float* __restrict__ C,          // M x N, fp32
    const float* __restrict__ Res,  // optional residual (M x N) or nullptr
    int M, int N, int K) {
    constexpr int BM = 128, BN = 128, BK = 32;
    __shared__ __align__(16) bf16_t sA[BM * BK];
    __shared__ __align__(16) bf16_t sB[BN * BK];

    const int tid = threadIdx.x;
    const int lane = tid & 31;
    const int wid = tid >> 5;
    const int waveM = wid & 3;   // 0..3 -> 32-row slice
    const int waveN = wid >> 2;  // 0..1 -> 64-col slice
    const int bm = blockIdx.y * BM;
    const int bn = blockIdx.x * BN;

    v8f acc[2][4] = {};

    // cooperative global->LDS mapping: 128 rows, 2 threads/row, 16 bf16 each
    const int grow = tid >> 1;
    const int gcol = (tid & 1) * 16;

    const int ksel = (lane & 16) ? 8 : 0;
    const int am = waveM * 32;
    const int bnl = waveN * 64;
    const int mlane = lane & 15;

    int nrow = bn + grow;
    if (nrow >= N) nrow = N - 1;  // clamp for the ragged last N-block

    const unsigned la = lds_off_of(&sA[grow * BK + gcol]);
    const unsigned lb = lds_off_of(&sB[grow * BK + gcol]);

    for (int k0 = 0; k0 < K; k0 += BK) {
        const bf16_t* ga = A + (size_t)(bm + grow) * K + k0 + gcol;
        const bf16_t* gb = Bt + (size_t)nrow * K + k0 + gcol;
        // async DMA: four 16B lane-transfers straight into LDS (no VGPR stage)
        async_load_b128(la, ga);
        async_load_b128(la + 16, ga + 8);
        async_load_b128(lb, gb);
        async_load_b128(lb + 16, gb + 8);

        if (k0 + BK < K) {  // prefetch next K tile (global_prefetch_b8)
            __builtin_prefetch(ga + BK, 0, 1);
            __builtin_prefetch(gb + BK, 0, 1);
        }
        wait_asynccnt0();
        __syncthreads();

        v16bf afrag[2], bfrag[4];
#pragma unroll
        for (int i = 0; i < 2; i++)
            afrag[i] = load_frag(&sA[(am + i * 16 + mlane) * BK + ksel]);
#pragma unroll
        for (int j = 0; j < 4; j++)
            bfrag[j] = load_frag(&sB[(bnl + j * 16 + mlane) * BK + ksel]);
#pragma unroll
        for (int i = 0; i < 2; i++)
#pragma unroll
            for (int j = 0; j < 4; j++)
                acc[i][j] = __builtin_amdgcn_wmma_f32_16x16x32_bf16(
                    false, afrag[i], false, bfrag[j], (short)0, acc[i][j],
                    false, false);
        __syncthreads();
    }

    const int moff = (lane & 16) ? 8 : 0;
#pragma unroll
    for (int i = 0; i < 2; i++) {
#pragma unroll
        for (int j = 0; j < 4; j++) {
            const int nglob = bn + bnl + j * 16 + mlane;
            if (nglob < N) {
                const int mbase = bm + am + i * 16 + moff;
#pragma unroll
                for (int r = 0; r < 8; r++) {
                    size_t idx = (size_t)(mbase + r) * N + nglob;
                    float val = acc[i][j][r];
                    if (Res) val += Res[idx];
                    C[idx] = val;
                }
            }
        }
    }
}

// ---------------------------------------------------------------------------
// Causal depthwise conv (width 4) + bias + SiLU over the xBC slice of zxbcdt.
// ---------------------------------------------------------------------------
__global__ void conv_silu_kernel(const float* __restrict__ zxbcdt,
                                 const float* __restrict__ cw,
                                 const float* __restrict__ cb,
                                 float* __restrict__ xconv) {
    int idx = blockIdx.x * blockDim.x + threadIdx.x;
    const int total = NTOK * CONV_DIM;
    if (idx >= total) return;
    const int c = idx % CONV_DIM;
    const int tt = idx / CONV_DIM;
    const int t = tt % SEQLEN;
    float s = cb[c];
#pragma unroll
    for (int j = 0; j < D_CONV; j++) {
        int tj = t - (D_CONV - 1) + j;
        if (tj >= 0) {
            size_t src = (size_t)(tt - (D_CONV - 1) + j) * D_IN_PROJ + D_INNER + c;
            s += zxbcdt[src] * cw[c * D_CONV + j];
        }
    }
    s = s / (1.f + __expf(-s));  // SiLU
    xconv[(size_t)tt * CONV_DIM + c] = s;
}

// ---------------------------------------------------------------------------
// dt = softplus(dt_raw + bias)
// ---------------------------------------------------------------------------
__global__ void dt_kernel(const float* __restrict__ zxbcdt,
                          const float* __restrict__ dt_bias,
                          float* __restrict__ dt_out) {
    int idx = blockIdx.x * blockDim.x + threadIdx.x;
    if (idx >= NTOK * NHEADS) return;
    const int h = idx % NHEADS;
    const int tt = idx / NHEADS;
    float x = zxbcdt[(size_t)tt * D_IN_PROJ + D_INNER + CONV_DIM + h] + dt_bias[h];
    dt_out[idx] = (x > 20.f) ? x : log1pf(__expf(x));
}

// ---------------------------------------------------------------------------
// Chunked SSD, phase A (intra-chunk).  One block per (b,chunk,h) tile.
//   La[t]   = cumsum(dt*A) within chunk (fp32)
//   G       = C @ B^T                       (WMMA)
//   Gs[t,s] = (s<=t) ? G * exp(La[t]-La[s]) * dt[s] : 0
//   Y_intra = Gs @ X  (+ D*x)               (WMMA)
//   S[n,p]  = sum_s exp(La[63]-La[s]) dt[s] B[s,n] x[s,p]   (WMMA)
// ---------------------------------------------------------------------------
__global__ __launch_bounds__(256) void chunk_intra_kernel(
    const float* __restrict__ xconv, const float* __restrict__ dtb,
    const float* __restrict__ A_log, const float* __restrict__ Dp,
    float* __restrict__ y, float* __restrict__ Schunk,
    float* __restrict__ La_buf) {
    __shared__ __align__(16) bf16_t sXt[64 * 64];  // [p][s]  (X transposed)
    __shared__ __align__(16) bf16_t sB[64 * 64];   // [s][n]
    __shared__ __align__(16) bf16_t sC[64 * 64];   // [t][n]
    __shared__ __align__(16) bf16_t sG[64 * 64];   // [t][s]  masked/scaled
    __shared__ __align__(16) bf16_t sBw[64 * 64];  // [n][s]  decayed B^T
    __shared__ float sdt[64], sLa[64];

    const int bc = blockIdx.x;            // 0..B*NCHUNK-1
    const int b = bc >> 6, cc = bc & 63;  // NCHUNK == 64
    const int h = blockIdx.y;
    const int tt0 = b * SEQLEN + cc * QCHUNK;
    const int tile = bc * NHEADS + h;
    const int tid = threadIdx.x;

    // ---- stage chunk operands into LDS (bf16) ----
    {
        const int s = tid >> 2;
        const int c16 = (tid & 3) * 16;
        const float* row = xconv + (size_t)(tt0 + s) * CONV_DIM;
#pragma unroll
        for (int i = 0; i < 16; i++) {
            const int p = c16 + i;
            sXt[p * 64 + s] = (bf16_t)row[h * HEADDIM + p];
            sB[s * 64 + p] = (bf16_t)row[D_INNER + p];
            sC[s * 64 + p] = (bf16_t)row[D_INNER + D_STATE + p];
        }
    }
    if (tid < 64) sdt[tid] = dtb[(size_t)(tt0 + tid) * NHEADS + h];
    __syncthreads();
    if (tid == 0) {
        const float Ah = -__expf(A_log[h]);
        float a = 0.f;
        for (int u = 0; u < 64; u++) {
            a += sdt[u] * Ah;
            sLa[u] = a;
        }
    }
    __syncthreads();
    if (tid < 64) La_buf[(size_t)(tt0 + tid) * NHEADS + h] = sLa[tid];

    const int lane = tid & 31, wid = tid >> 5;
    const int trow = (wid >> 1) * 16 + ((lane & 16) ? 8 : 0);
    const int ncol0 = (wid & 1) * 32;
    const int mlane = lane & 15;

    // ---- GEMM1: G = C @ B^T ----
    v8f accG[2] = {};
    gemm64_accum(sC, sB, lane, wid, accG);

    // mask + decay + dt scale -> sG (bf16)
#pragma unroll
    for (int j = 0; j < 2; j++) {
        const int scol = ncol0 + j * 16 + mlane;
#pragma unroll
        for (int r = 0; r < 8; r++) {
            const int t = trow + r;
            float g = (scol <= t)
                          ? accG[j][r] * __expf(sLa[t] - sLa[scol]) * sdt[scol]
                          : 0.f;
            sG[t * 64 + scol] = (bf16_t)g;
        }
    }
    // decayed B^T for the chunk-state GEMM
    {
        const float LaQ = sLa[63];
        for (int e = tid; e < 64 * 64; e += 256) {
            const int n = e >> 6, ss = e & 63;
            const float wgt = __expf(LaQ - sLa[ss]) * sdt[ss];
            sBw[n * 64 + ss] = (bf16_t)((float)sB[ss * 64 + n] * wgt);
        }
    }
    __syncthreads();

    // ---- GEMM2: Y_intra = Gs @ X    ---- (Bt = Xt)
    // ---- GEMM3: S       = Bw @ X    ---- (Bt = Xt)
    v8f accY[2] = {}, accS[2] = {};
    gemm64_accum(sG, sXt, lane, wid, accY);
    gemm64_accum(sBw, sXt, lane, wid, accS);

    const float Dh = Dp[h];
#pragma unroll
    for (int j = 0; j < 2; j++) {
        const int pcol = ncol0 + j * 16 + mlane;
#pragma unroll
        for (int r = 0; r < 8; r++) {
            const int t = trow + r;  // row = t for Y, row = n for S
            float yv = accY[j][r] + Dh * (float)sXt[pcol * 64 + t];
            y[(size_t)(tt0 + t) * D_INNER + h * HEADDIM + pcol] = yv;
            Schunk[(size_t)tile * 4096 + t * 64 + pcol] = accS[j][r];
        }
    }
}

// ---------------------------------------------------------------------------
// Chunked SSD, phase B: sequential carry over 64 chunks per (b,h).
//   For each chunk: emit R_prev (transposed, bf16), then
//   R = exp(La_chunk_end) * R + S_chunk.
// ---------------------------------------------------------------------------
__global__ __launch_bounds__(256) void chunk_carry_kernel(
    const float* __restrict__ Schunk, const float* __restrict__ La_buf,
    bf16_t* __restrict__ Rt_bf) {
    const int bh = blockIdx.x;  // B*NHEADS = 64
    const int b = bh >> 5, h = bh & 31;
    const int tid = threadIdx.x;
    float R[16];
#pragma unroll
    for (int i = 0; i < 16; i++) R[i] = 0.f;

    for (int cc = 0; cc < NCHUNK; cc++) {
        const int tile = (b * NCHUNK + cc) * NHEADS + h;
#pragma unroll
        for (int i = 0; i < 16; i++) {
            const int e = tid * 16 + i;  // e = n*64 + p
            const int n = e >> 6, p = e & 63;
            Rt_bf[(size_t)tile * 4096 + p * 64 + n] = (bf16_t)R[i];
        }
        const float decay =
            __expf(La_buf[(size_t)(b * SEQLEN + cc * QCHUNK + 63) * NHEADS + h]);
#pragma unroll
        for (int i = 0; i < 16; i++) {
            const int e = tid * 16 + i;
            R[i] = decay * R[i] + Schunk[(size_t)tile * 4096 + e];
        }
    }
}

// ---------------------------------------------------------------------------
// Chunked SSD, phase C (inter-chunk): y += exp(La[t]) * (C @ R_prev)   (WMMA)
// R_prev tile fetched with async global->LDS DMA (pure 8KB bf16 copy).
// ---------------------------------------------------------------------------
__global__ __launch_bounds__(256) void chunk_inter_kernel(
    const float* __restrict__ xconv, const float* __restrict__ La_buf,
    const bf16_t* __restrict__ Rt_bf, float* __restrict__ y) {
    __shared__ __align__(16) bf16_t sC[64 * 64];   // [t][n]
    __shared__ __align__(16) bf16_t sRt[64 * 64];  // [p][n]
    __shared__ float sLa[64];

    const int bc = blockIdx.x;
    const int b = bc >> 6, cc = bc & 63;
    const int h = blockIdx.y;
    const int tt0 = b * SEQLEN + cc * QCHUNK;
    const int tile = bc * NHEADS + h;
    const int tid = threadIdx.x;

    // async DMA the R_prev tile into LDS while we convert C
    {
        const bf16_t* src = Rt_bf + (size_t)tile * 4096 + tid * 16;
        const unsigned lr = lds_off_of(&sRt[tid * 16]);
        async_load_b128(lr, src);
        async_load_b128(lr + 16, src + 8);
    }
    {
        const int s = tid >> 2;
        const int c16 = (tid & 3) * 16;
        const float* row = xconv + (size_t)(tt0 + s) * CONV_DIM;
#pragma unroll
        for (int i = 0; i < 16; i++) {
            const int n = c16 + i;
            sC[s * 64 + n] = (bf16_t)row[D_INNER + D_STATE + n];
        }
    }
    if (tid < 64) sLa[tid] = La_buf[(size_t)(tt0 + tid) * NHEADS + h];
    wait_asynccnt0();
    __syncthreads();

    const int lane = tid & 31, wid = tid >> 5;
    v8f acc[2] = {};
    gemm64_accum(sC, sRt, lane, wid, acc);

    const int trow = (wid >> 1) * 16 + ((lane & 16) ? 8 : 0);
    const int ncol0 = (wid & 1) * 32;
#pragma unroll
    for (int j = 0; j < 2; j++) {
        const int p = ncol0 + j * 16 + (lane & 15);
#pragma unroll
        for (int r = 0; r < 8; r++) {
            const int t = trow + r;
            const size_t idx = (size_t)(tt0 + t) * D_INNER + h * HEADDIM + p;
            y[idx] += __expf(sLa[t]) * acc[j][r];
        }
    }
}

// ---------------------------------------------------------------------------
// y = y * silu(z); gated RMSNorm; emit bf16 for the out_proj WMMA GEMM.
// ---------------------------------------------------------------------------
__global__ __launch_bounds__(256) void gate_rmsnorm_kernel(
    const float* __restrict__ yin, const float* __restrict__ zxbcdt,
    const float* __restrict__ rmsw, bf16_t* __restrict__ yout) {
    __shared__ float sred[8];
    __shared__ float sscale;
    const int tt = blockIdx.x;
    const float* yr = yin + (size_t)tt * D_INNER;
    const float* zr = zxbcdt + (size_t)tt * D_IN_PROJ;
    float v[8];
    float ss = 0.f;
#pragma unroll
    for (int i = 0; i < 8; i++) {
        int j = threadIdx.x + 256 * i;
        float z = zr[j];
        float g = z / (1.f + __expf(-z));
        float yv = yr[j] * g;
        v[i] = yv;
        ss += yv * yv;
    }
#pragma unroll
    for (int o = 16; o > 0; o >>= 1) ss += __shfl_down(ss, o);
    if ((threadIdx.x & 31) == 0) sred[threadIdx.x >> 5] = ss;
    __syncthreads();
    if (threadIdx.x == 0) {
        float tot = 0.f;
        for (int i = 0; i < 8; i++) tot += sred[i];
        sscale = rsqrtf(tot * (1.f / D_INNER) + EPS_F);
    }
    __syncthreads();
    const float sc = sscale;
#pragma unroll
    for (int i = 0; i < 8; i++) {
        int j = threadIdx.x + 256 * i;
        yout[(size_t)tt * D_INNER + j] = (bf16_t)(v[i] * sc * rmsw[j]);
    }
}

// ---------------------------------------------------------------------------
// Orchestration
// ---------------------------------------------------------------------------
extern "C" void kernel_launch(void* const* d_in, const int* in_sizes, int n_in,
                              void* d_out, int out_size, void* d_ws,
                              size_t ws_size, hipStream_t stream) {
    const float* x       = (const float*)d_in[0];
    const float* ln_w    = (const float*)d_in[1];
    const float* ln_b    = (const float*)d_in[2];
    const float* in_w    = (const float*)d_in[3];
    const float* conv_w  = (const float*)d_in[4];
    const float* conv_b  = (const float*)d_in[5];
    const float* dt_bias = (const float*)d_in[6];
    const float* A_log   = (const float*)d_in[7];
    const float* Dp      = (const float*)d_in[8];
    const float* rms_w   = (const float*)d_in[9];
    const float* out_w   = (const float*)d_in[10];
    float* out = (float*)d_out;

    char* ws = (char*)d_ws;
    size_t off = 0;
    auto alloc = [&](size_t bytes) -> char* {
        char* p = ws + off;
        off += (bytes + 255) & ~(size_t)255;
        return p;
    };
    bf16_t* w_in_bf  = (bf16_t*)alloc((size_t)NLAYERS * D_IN_PROJ * D_MODEL * 2);
    bf16_t* w_out_bf = (bf16_t*)alloc((size_t)NLAYERS * D_MODEL * D_INNER * 2);
    bf16_t* a_bf     = (bf16_t*)alloc((size_t)NTOK * D_MODEL * 2);
    bf16_t* yg_bf    = (bf16_t*)alloc((size_t)NTOK * D_INNER * 2);
    bf16_t* Rt_bf    = (bf16_t*)alloc((size_t)NTILES * 64 * 64 * 2);
    float* zxbcdt    = (float*)alloc((size_t)NTOK * D_IN_PROJ * 4);
    float* xconv     = (float*)alloc((size_t)NTOK * CONV_DIM * 4);
    float* dtb       = (float*)alloc((size_t)NTOK * NHEADS * 4);
    float* La_buf    = (float*)alloc((size_t)NTOK * NHEADS * 4);
    float* Schunk    = (float*)alloc((size_t)NTILES * 64 * 64 * 4);
    float* ybuf      = (float*)alloc((size_t)NTOK * D_INNER * 4);
    float* h1        = (float*)alloc((size_t)NTOK * D_MODEL * 4);

    {
        int n = NLAYERS * D_IN_PROJ * D_MODEL;
        f32_to_bf16_kernel<<<(n + 1023) / 1024, 256, 0, stream>>>(in_w, w_in_bf, n);
        int m = NLAYERS * D_MODEL * D_INNER;
        f32_to_bf16_kernel<<<(m + 1023) / 1024, 256, 0, stream>>>(out_w, w_out_bf, m);
    }

    const float* hin = x;
    for (int l = 0; l < NLAYERS; l++) {
        const bf16_t* wi = w_in_bf + (size_t)l * D_IN_PROJ * D_MODEL;
        const bf16_t* wo = w_out_bf + (size_t)l * D_MODEL * D_INNER;

        layernorm_bf16_kernel<<<NTOK, 256, 0, stream>>>(
            hin, ln_w + l * D_MODEL, ln_b + l * D_MODEL, a_bf);

        dim3 g1((D_IN_PROJ + 127) / 128, NTOK / 128);
        gemm_bf16_wmma_kernel<<<g1, 256, 0, stream>>>(
            a_bf, wi, zxbcdt, nullptr, NTOK, D_IN_PROJ, D_MODEL);

        int nc = NTOK * CONV_DIM;
        conv_silu_kernel<<<(nc + 255) / 256, 256, 0, stream>>>(
            zxbcdt, conv_w + l * CONV_DIM * D_CONV, conv_b + l * CONV_DIM, xconv);

        int nd = NTOK * NHEADS;
        dt_kernel<<<(nd + 255) / 256, 256, 0, stream>>>(
            zxbcdt, dt_bias + l * NHEADS, dtb);

        dim3 gch(BATCH * NCHUNK, NHEADS);
        chunk_intra_kernel<<<gch, 256, 0, stream>>>(
            xconv, dtb, A_log + l * NHEADS, Dp + l * NHEADS, ybuf, Schunk, La_buf);

        chunk_carry_kernel<<<BATCH * NHEADS, 256, 0, stream>>>(
            Schunk, La_buf, Rt_bf);

        chunk_inter_kernel<<<gch, 256, 0, stream>>>(xconv, La_buf, Rt_bf, ybuf);

        gate_rmsnorm_kernel<<<NTOK, 256, 0, stream>>>(
            ybuf, zxbcdt, rms_w + l * D_INNER, yg_bf);

        float* dest = (l == NLAYERS - 1) ? out : h1;
        dim3 g2(D_MODEL / 128, NTOK / 128);
        gemm_bf16_wmma_kernel<<<g2, 256, 0, stream>>>(
            yg_bf, wo, dest, hin, NTOK, D_MODEL, D_INNER);
        hin = h1;
    }
    (void)in_sizes; (void)n_in; (void)out_size; (void)ws_size;
}
